// RadialFormerLayer_69312182223503
// MI455X (gfx1250) — matrix operand order repaired and checked by reading, not compiled, not run on hardware
//
#include <hip/hip_runtime.h>
#include <hip/hip_bf16.h>

#define BB 4
#define NN 16384
#define DD 512
#define HH 8
#define HD 64

typedef __attribute__((ext_vector_type(16))) _Float16 v16h;
typedef __attribute__((ext_vector_type(8)))  _Float16 v8h;
typedef __attribute__((ext_vector_type(4)))  _Float16 v4h;
typedef __attribute__((ext_vector_type(8)))  float    v8f;
typedef __attribute__((ext_vector_type(4)))  unsigned int u32x4;
typedef __attribute__((ext_vector_type(8)))  int      i32x8;
typedef __attribute__((ext_vector_type(4)))  int      i32x4;

// ---------------------------------------------------------------- WMMA helpers

__device__ inline v8f wmma32(v16h a, v16h b, v8f c) {
  // v_wmma_f32_16x16x32_f16: (neg_a, A, neg_b, B, c_mod, C, reuse_a, reuse_b)
  return __builtin_amdgcn_wmma_f32_16x16x32_f16(false, a, false, b, (short)0, c,
                                                false, false);
}

// A-fragment (16x32 MxK, f16) from an LDS f16 tile [16][512].
// Lane L: m = L&15, K runs [kb, kb+8) and [kb+16, kb+24), kb = k0 + (L>>4)*8.
__device__ inline v16h load_a_l(const _Float16* A, int k0, int lane) {
  const int m  = lane & 15;
  const int kb = k0 + ((lane >> 4) << 3);
  v8h lo = *(const v8h*)(A + m * DD + kb);
  v8h hi = *(const v8h*)(A + m * DD + kb + 16);
  v16h a;
#pragma unroll
  for (int i = 0; i < 8; ++i) { a[i] = lo[i]; a[8 + i] = hi[i]; }
  return a;
}

// B-fragment (32x16 KxN) from pre-transposed f16 weight Wt[k][n] (ld=512).
// Lane L: n = n0 + (L&15), K runs [kb, kb+8) and [kb+16, kb+24), kb = k0 + (L>>4)*8.
__device__ inline v16h load_b(const _Float16* __restrict__ Wt, int k0, int n0, int lane) {
  const _Float16* p = Wt + (size_t)(k0 + ((lane >> 4) << 3)) * DD + n0 + (lane & 15);
  v16h b;
#pragma unroll
  for (int i = 0; i < 8; ++i) { b[i] = p[(size_t)i * DD]; b[8 + i] = p[(size_t)(16 + i) * DD]; }
  return b;
}

// ------------------------------------------------------- TDM tile staging
// Stage a 16-row x 512-col f32 tile (32 KB) from global into LDS with the
// Tensor Data Mover.  D# per CDNA5 ISA ch.8: group0={count=1, lds_addr,
// global_addr, type=2}; group1={mask=0, data_size=4B, tensor_dim0=512,
// tensor_dim1=Mtot, tile_dim0=512, tile_dim1=16, dim0_stride=512}.
// Caller: one wave only; follows with s_wait_tensorcnt 0 + barrier.
__device__ inline void tdm_load_tile_f32(const float* gbase, unsigned lds_off,
                                         int row0, int Mtot) {
  unsigned long long ga =
      (unsigned long long)(size_t)gbase + (unsigned long long)row0 * (DD * 4ull);
  u32x4 g0;
  g0[0] = 1u;                                             // count=1, user desc
  g0[1] = lds_off;                                        // LDS byte address
  g0[2] = (unsigned)ga;                                   // global_addr[31:0]
  g0[3] = (unsigned)((ga >> 32) & 0x01FFFFFFu) | 0x80000000u;  // addr[56:32]|type=2
  i32x8 g1;
  g1[0] = 0x00020000;                                     // data_size=4B, mask=0
  g1[1] = (DD & 0xFFFF) << 16;                            // tensor_dim0 lo16
  g1[2] = (Mtot & 0xFFFF) << 16;                          // dim0 hi=0 | dim1 lo16
  g1[3] = ((Mtot >> 16) & 0xFFFF) | ((DD & 0xFFFF) << 16);// dim1 hi | tile_dim0=512
  g1[4] = 16;                                             // tile_dim1=16, tile_dim2=0
  g1[5] = DD;                                             // dim0_stride = 512
  g1[6] = 0;
  g1[7] = 0;
  i32x4 z4 = {0, 0, 0, 0};
  i32x8 z8 = {0, 0, 0, 0, 0, 0, 0, 0};
  __builtin_amdgcn_tensor_load_to_lds(g0, g1, z4, z4, z8, 0);
}

// Cooperative f32 -> f16 conversion of the staged tile (all 256 threads).
__device__ inline void convert_tile(const float* __restrict__ fX,
                                    _Float16* __restrict__ sX, int t) {
  const float4* src = (const float4*)fX;
  for (int c = t; c < (16 * DD) / 4; c += 256) {
    float4 v = src[c];
    v4h h;
    h[0] = (_Float16)v.x; h[1] = (_Float16)v.y;
    h[2] = (_Float16)v.z; h[3] = (_Float16)v.w;
    *(v4h*)(sX + 4 * c) = h;
  }
}

// ---------------------------------------------------------------- weight convert
// Out[mat][k*512+n] = (f16) In[mat][n*512+k]   (transpose to [K][N])
__global__ __launch_bounds__(256) void wconv(const float* __restrict__ A0,
                                             const float* __restrict__ A1,
                                             const float* __restrict__ A2,
                                             const float* __restrict__ A3,
                                             _Float16* __restrict__ O) {
  int idx = blockIdx.x * 256 + threadIdx.x;        // 4 * 512*512 total
  int mat = idx >> 18;
  int rem = idx & 262143;
  int k = rem >> 9, n = rem & 511;
  const float* src = (mat == 0) ? A0 : (mat == 1) ? A1 : (mat == 2) ? A2 : A3;
  O[idx] = (_Float16)src[n * DD + k];
}

// ---------------------------------------------------------------- r projections
__global__ __launch_bounds__(256) void rproj(const float* __restrict__ r,
                                             const float* __restrict__ Wq,
                                             const float* __restrict__ Wk,
                                             const float* __restrict__ Wv,
                                             float* __restrict__ q2,
                                             float* __restrict__ rK,
                                             float* __restrict__ rV) {
  int idx = blockIdx.x * 256 + threadIdx.x;        // B*512
  int b = idx >> 9, n = idx & 511;
  const float* rb = r + b * DD;
  const float* wq = Wq + (size_t)n * DD;
  const float* wk = Wk + (size_t)n * DD;
  const float* wv = Wv + (size_t)n * DD;
  float aq = 0.f, ak = 0.f, av = 0.f;
  for (int k = 0; k < DD; ++k) {
    float x = rb[k];
    aq += x * wq[k]; ak += x * wk[k]; av += x * wv[k];
  }
  q2[idx] = aq; rK[idx] = ak; rV[idx] = av;
}

// ---------------------------------------------------------------- dual GEMM
// OutA = X WtA, OutB = X WtB for one 32-row tile of X (TDM-staged in two
// 16-row chunks through a 32 KB bounce buffer; 64 KB static LDS total).
// WG = 8 waves; 16 jobs of 32x64 (2 mats x 8 col-blocks); each B fragment
// feeds two WMMAs (rows 0-15 and 16-31) -> half the weight fetch traffic.
template <bool OUT_F16>
__global__ __launch_bounds__(256) void gemm2(const float* __restrict__ X,
                                             const _Float16* __restrict__ WtA,
                                             const _Float16* __restrict__ WtB,
                                             void* __restrict__ OutA,
                                             void* __restrict__ OutB,
                                             int Mtot) {
  __shared__ __align__(16) float    fX[16 * DD];     // 32 KB staging (reused 2x)
  __shared__ __align__(16) _Float16 sX[32 * DD];     // 32 KB f16 tile (32 rows)
  const int t = threadIdx.x, lane = t & 31, wv = t >> 5;
  const int row0 = blockIdx.x * 32;

  for (int c = 0; c < 2; ++c) {
    if (wv == 0) {
      tdm_load_tile_f32(X, (unsigned)(size_t)(void*)fX, row0 + 16 * c, Mtot);
      __builtin_amdgcn_s_wait_tensorcnt(0);
    }
    __syncthreads();
    convert_tile(fX, sX + c * 16 * DD, t);
    __syncthreads();                                 // fX reusable / sX ready
  }

  for (int j = wv; j < 16; j += 8) {
    const _Float16* W = (j < 8) ? WtA : WtB;
    void* Out = (j < 8) ? OutA : OutB;
    const int n0 = (j & 7) * 64;
    v8f acc0[4] = {}, acc1[4] = {};
    for (int k0 = 0; k0 < DD; k0 += 32) {
      v16h a0 = load_a_l(sX, k0, lane);              // rows 0-15
      v16h a1 = load_a_l(sX + 16 * DD, k0, lane);    // rows 16-31
#pragma unroll
      for (int u = 0; u < 4; ++u) {
        v16h bf = load_b(W, k0, n0 + 16 * u, lane);
        acc0[u] = wmma32(a0, bf, acc0[u]);
        acc1[u] = wmma32(a1, bf, acc1[u]);
      }
    }
    const int cn = lane & 15, rb = (lane >> 4) << 3;
#pragma unroll
    for (int u = 0; u < 4; ++u)
#pragma unroll
      for (int v = 0; v < 8; ++v) {
        const size_t i0 = (size_t)(row0 + rb + v) * DD + n0 + 16 * u + cn;
        const size_t i1 = i0 + (size_t)16 * DD;
        if constexpr (OUT_F16) {
          ((_Float16*)Out)[i0] = (_Float16)acc0[u][v];
          ((_Float16*)Out)[i1] = (_Float16)acc1[u][v];
        } else {
          ((float*)Out)[i0] = acc0[u][v];
          ((float*)Out)[i1] = acc1[u][v];
        }
      }
  }
}

// ---------------------------------------------------------------- fused stage 1
// Per 16-row block of S (flattened [B*N,512]):
//   TDM-stage X tile -> convert f16
//   A) Q,K,V = X * {Wq,Wk,Wv}^T          (WMMA, LDS f16 tiles)
//   B) 3-token softmax per (row, head)   (attended value overwrites Q tile)
//   C) OUT = A * Wo^T, ReLU              (WMMA, f32 LDS overlays K,V)
//   D) LayerNorm per row -> S_new
// LDS: [Q 16K | K 16K | V 16K | X16 16K] = 64 KB; f32 staging overlays Q+K.
__global__ __launch_bounds__(256) void stage1(const float* __restrict__ S,
                                              const float* __restrict__ MK,
                                              const float* __restrict__ MV,
                                              const float* __restrict__ rK,
                                              const float* __restrict__ rV,
                                              const _Float16* __restrict__ Wq16,
                                              const _Float16* __restrict__ Wk16,
                                              const _Float16* __restrict__ Wv16,
                                              const _Float16* __restrict__ Wo16,
                                              const float* __restrict__ gs,
                                              const float* __restrict__ bs,
                                              float* __restrict__ Snew) {
  __shared__ __align__(16) _Float16 smem[4 * 16 * DD];   // 64 KB
  _Float16* sQ = smem;                                   // Q, later A
  _Float16* sX = smem + 3 * 16 * DD;                     // f16 X tile
  float*    fX = (float*)smem;                           // f32 staging (Q+K slots)
  float* OUTBUF = (float*)(smem + 16 * DD);              // phase C/D (K+V slots)

  const int t = threadIdx.x, lane = t & 31, wv = t >> 5;
  const int row0 = blockIdx.x * 16;                      // row in [0, B*N)

  if (wv == 0) {
    tdm_load_tile_f32(S, (unsigned)(size_t)(void*)fX, row0, BB * NN);
    __builtin_amdgcn_s_wait_tensorcnt(0);
  }
  __syncthreads();
  convert_tile(fX, sX, t);
  __syncthreads();                                       // fX dead after this

  // ---- Phase A: QKV projection (24 jobs of 16x64 over 8 waves)
  for (int j = wv; j < 24; j += 8) {
    const int mat = j >> 3;                // 0=Q 1=K 2=V
    const int n0 = (j & 7) * 64;
    const _Float16* W = (mat == 0) ? Wq16 : (mat == 1) ? Wk16 : Wv16;
    _Float16* dst = smem + mat * 16 * DD;
    v8f acc[4] = {};
    for (int k0 = 0; k0 < DD; k0 += 32) {
      v16h a = load_a_l(sX, k0, lane);
#pragma unroll
      for (int u = 0; u < 4; ++u) {
        v16h bf = load_b(W, k0, n0 + 16 * u, lane);
        acc[u] = wmma32(a, bf, acc[u]);
      }
    }
    const int cn = lane & 15, rb = (lane >> 4) << 3;
#pragma unroll
    for (int u = 0; u < 4; ++u)
#pragma unroll
      for (int v = 0; v < 8; ++v)
        dst[(rb + v) * DD + n0 + 16 * u + cn] = (_Float16)acc[u][v];
  }
  __syncthreads();

  // ---- Phase B: 3-token attention (threads 0..127 -> 16 rows x 8 heads)
  if (t < 128) {
    const int rowl = t >> 3, h = t & 7;
    const int g = row0 + rowl;
    const int b = g >> 14;                 // / N
    const int i = g & (NN - 1);            // % N
    _Float16* Qh = sQ + rowl * DD + h * HD;
    const _Float16* Kh = smem + 16 * DD + rowl * DD + h * HD;
    const _Float16* Vh = smem + 32 * DD + rowl * DD + h * HD;
    const float* mk = MK + (size_t)i * DD + h * HD;
    const float* mv = MV + (size_t)i * DD + h * HD;
    const float* rk = rK + b * DD + h * HD;
    const float* rv = rV + b * DD + h * HD;
    float s0 = 0.f, s1 = 0.f, s2 = 0.f;
    for (int d = 0; d < HD; ++d) {
      float q = (float)Qh[d];
      s0 += q * (float)Kh[d]; s1 += q * mk[d]; s2 += q * rk[d];
    }
    s0 *= 0.125f; s1 *= 0.125f; s2 *= 0.125f;       // 1/sqrt(64)
    float mm = fmaxf(s0, fmaxf(s1, s2));
    float e0 = __expf(s0 - mm), e1 = __expf(s1 - mm), e2 = __expf(s2 - mm);
    float inv = 1.0f / (e0 + e1 + e2);
    e0 *= inv; e1 *= inv; e2 *= inv;
    for (int d = 0; d < HD; ++d)
      Qh[d] = (_Float16)(e0 * (float)Vh[d] + e1 * mv[d] + e2 * rv[d]);
  }
  __syncthreads();

  // ---- Phase C: OUT = A * Wo^T, ReLU
  {
    const int n0 = wv * 64;
    v8f acc[4] = {};
    for (int k0 = 0; k0 < DD; k0 += 32) {
      v16h a = load_a_l(sQ, k0, lane);
#pragma unroll
      for (int u = 0; u < 4; ++u) {
        v16h bf = load_b(Wo16, k0, n0 + 16 * u, lane);
        acc[u] = wmma32(a, bf, acc[u]);
      }
    }
    const int cn = lane & 15, rb = (lane >> 4) << 3;
#pragma unroll
    for (int u = 0; u < 4; ++u)
#pragma unroll
      for (int v = 0; v < 8; ++v)
        OUTBUF[(rb + v) * DD + n0 + 16 * u + cn] = fmaxf(acc[u][v], 0.0f);
  }
  __syncthreads();

  // ---- Phase D: LayerNorm, wave w handles rows 2w, 2w+1
  for (int rr = 0; rr < 2; ++rr) {
    const int rowl = wv * 2 + rr;
    const float* x = OUTBUF + rowl * DD;
    float s = 0.f, sq = 0.f;
    for (int c = lane; c < DD; c += 32) { float v = x[c]; s += v; sq += v * v; }
    for (int off = 16; off; off >>= 1) { s += __shfl_xor(s, off); sq += __shfl_xor(sq, off); }
    float mu = s * (1.0f / DD);
    float var = sq * (1.0f / DD) - mu * mu;
    float rs = rsqrtf(var + 1e-5f);
    float* yo = Snew + (size_t)(row0 + rowl) * DD;
    for (int c = lane; c < DD; c += 32)
      yo[c] = (x[c] - mu) * rs * gs[c] + bs[c];
  }
}

// ---------------------------------------------------------------- stage 2 attn
// One WG per (b,h). 8 warps stream tokens with online softmax; token 0 is r.
__global__ __launch_bounds__(256) void attn2(const float* __restrict__ q2,
                                             const float* __restrict__ rK,
                                             const float* __restrict__ rV,
                                             const _Float16* __restrict__ K2,
                                             const _Float16* __restrict__ V2,
                                             float* __restrict__ O2) {
  __shared__ float wM[8], wS[8], wA[8][HD];
  const int b = blockIdx.x >> 3, h = blockIdx.x & 7;
  const int lane = threadIdx.x & 31, wv = threadIdx.x >> 5;
  const float* qh = q2 + b * DD + h * HD;
  const float qa = qh[lane], qb = qh[lane + 32];
  float Mx = -3.0e38f, Ss = 0.f, a0 = 0.f, a1 = 0.f;
  if (wv == 0) {                               // token 0 = r itself
    const float* kh = rK + b * DD + h * HD;
    float d = qa * kh[lane] + qb * kh[lane + 32];
    for (int off = 16; off; off >>= 1) d += __shfl_xor(d, off);
    Mx = d * 0.125f; Ss = 1.0f;
    const float* vh = rV + b * DD + h * HD;
    a0 = vh[lane]; a1 = vh[lane + 32];
  }
  const size_t base = (size_t)b * NN * DD + h * HD;
  for (int l = wv; l < NN; l += 8) {
    const _Float16* kp = K2 + base + (size_t)l * DD;
    const _Float16* vp = V2 + base + (size_t)l * DD;
    __builtin_prefetch(kp + 8 * DD, 0, 0);      // global_prefetch_b8, next token
    float d = qa * (float)kp[lane] + qb * (float)kp[lane + 32];
    for (int off = 16; off; off >>= 1) d += __shfl_xor(d, off);
    d *= 0.125f;
    float Mn = fmaxf(Mx, d);
    float corr = __expf(Mx - Mn);
    float e = __expf(d - Mn);
    Ss = Ss * corr + e;
    a0 = a0 * corr + e * (float)vp[lane];
    a1 = a1 * corr + e * (float)vp[lane + 32];
    Mx = Mn;
  }
  if (lane == 0) { wM[wv] = Mx; wS[wv] = Ss; }
  wA[wv][lane] = a0; wA[wv][lane + 32] = a1;
  __syncthreads();
  if (wv == 0) {
    float Mf = wM[0];
    for (int w = 1; w < 8; ++w) Mf = fmaxf(Mf, wM[w]);
    float T = 0.f, o0 = 0.f, o1 = 0.f;
    for (int w = 0; w < 8; ++w) {
      float sc = __expf(wM[w] - Mf);
      T += wS[w] * sc; o0 += wA[w][lane] * sc; o1 += wA[w][lane + 32] * sc;
    }
    float inv = 1.0f / T;
    O2[b * DD + h * HD + lane] = o0 * inv;
    O2[b * DD + h * HD + lane + 32] = o1 * inv;
  }
}

// ---------------------------------------------------------------- final r_upd
__global__ __launch_bounds__(256) void finr(const float* __restrict__ O2,
                                            const float* __restrict__ Wo,
                                            const float* __restrict__ gr,
                                            const float* __restrict__ br,
                                            float* __restrict__ rout) {
  __shared__ float red1[256], red2[256], rowbuf[DD];
  const int b = blockIdx.x, t = threadIdx.x;
  const float* x = O2 + b * DD;
  float ls = 0.f, lq = 0.f;
  for (int c = t; c < DD; c += 256) {
    const float* w = Wo + (size_t)c * DD;
    float acc = 0.f;
    for (int k = 0; k < DD; ++k) acc += x[k] * w[k];
    acc = fmaxf(acc, 0.f);
    rowbuf[c] = acc;
    ls += acc; lq += acc * acc;
  }
  red1[t] = ls; red2[t] = lq;
  __syncthreads();
  for (int off = 128; off; off >>= 1) {
    if (t < off) { red1[t] += red1[t + off]; red2[t] += red2[t + off]; }
    __syncthreads();
  }
  float mu = red1[0] * (1.0f / DD);
  float var = red2[0] * (1.0f / DD) - mu * mu;
  float rs = rsqrtf(var + 1e-5f);
  for (int c = t; c < DD; c += 256)
    rout[b * DD + c] = (rowbuf[c] - mu) * rs * gr[c] + br[c];
}

// ---------------------------------------------------------------- launch
extern "C" void kernel_launch(void* const* d_in, const int* in_sizes, int n_in,
                              void* d_out, int out_size, void* d_ws, size_t ws_size,
                              hipStream_t stream) {
  (void)in_sizes; (void)n_in; (void)out_size; (void)ws_size;
  const float* r  = (const float*)d_in[0];
  const float* S  = (const float*)d_in[1];
  const float* M  = (const float*)d_in[2];
  const float* Wq = (const float*)d_in[3];
  const float* Wk = (const float*)d_in[4];
  const float* Wv = (const float*)d_in[5];
  const float* Wo = (const float*)d_in[6];
  const float* gs = (const float*)d_in[7];
  const float* bs = (const float*)d_in[8];
  const float* gr = (const float*)d_in[9];
  const float* br = (const float*)d_in[10];

  float* out   = (float*)d_out;
  float* r_out = out;                  // [B,512]
  float* Snew  = out + BB * DD;        // [B,N,512]

  // ---- workspace layout
  char* ws = (char*)d_ws;
  _Float16* w16  = (_Float16*)ws;                       // 4 x 512x512 f16 (2 MB)
  _Float16* Wq16 = w16;
  _Float16* Wk16 = w16 + 1 * DD * DD;
  _Float16* Wv16 = w16 + 2 * DD * DD;
  _Float16* Wo16 = w16 + 3 * DD * DD;
  float* f  = (float*)(ws + (size_t)4 * DD * DD * sizeof(_Float16));
  float* MK = f;                                        // [N,512] f32
  float* MV = MK + (size_t)NN * DD;                     // [N,512] f32
  float* rK = MV + (size_t)NN * DD;                     // [B,512]
  float* rV = rK + BB * DD;
  float* q2 = rV + BB * DD;
  float* O2 = q2 + BB * DD;                             // [B,512]
  _Float16* K2 = (_Float16*)(O2 + BB * DD);             // [B*N,512] f16
  _Float16* V2 = K2 + (size_t)BB * NN * DD;             // [B*N,512] f16

  // 1) weights -> f16, transposed to [K][N]
  wconv<<<4096, 256, 0, stream>>>(Wq, Wk, Wv, Wo, w16);
  // 2) r projections (f32, tiny)
  rproj<<<(BB * DD) / 256, 256, 0, stream>>>(r, Wq, Wk, Wv, q2, rK, rV);
  // 3) MK/MV = M {Wk,Wv}^T (TDM-staged, WMMA, single read of M)
  gemm2<false><<<NN / 32, 256, 0, stream>>>(M, Wk16, Wv16, MK, MV, NN);
  // 4) fused stage 1 -> S_new
  stage1<<<(BB * NN) / 16, 256, 0, stream>>>(S, MK, MV, rK, rV,
                                             Wq16, Wk16, Wv16, Wo16, gs, bs, Snew);
  // 5) K2/V2 = S_new {Wk,Wv}^T (f16 out, single read of S_new)
  gemm2<true><<<(BB * NN) / 32, 256, 0, stream>>>(Snew, Wk16, Wv16, K2, V2, BB * NN);
  // 6) radial attention over [r, S_new]
  attn2<<<BB * HH, 256, 0, stream>>>(q2, rK, rV, K2, V2, O2);
  // 7) r_upd = LN(relu(O2 Wo^T))
  finr<<<BB, 256, 0, stream>>>(O2, Wo, gr, br, r_out);
}
